// HunyuanSparseMoeBlock_53309134078388
// MI455X (gfx1250) — matrix operand order repaired and checked by reading, not compiled, not run on hardware
//
#include <hip/hip_runtime.h>

// ---------------------------------------------------------------------------
// Types
// ---------------------------------------------------------------------------
typedef __attribute__((ext_vector_type(16))) __bf16 v16bf;
typedef __attribute__((ext_vector_type(8)))  float  v8f;
typedef __attribute__((ext_vector_type(4)))  float  f32x4;
typedef __attribute__((ext_vector_type(4)))  unsigned int u32x4;

#define T_TOK 2048
#define HDIM  2048
#define NEXP  8

// ---------------------------------------------------------------------------
// fp32 <-> bf16 split helpers (round-to-nearest-even)
// ---------------------------------------------------------------------------
__device__ __forceinline__ unsigned short f32_to_bf16(float x) {
    unsigned int u = __builtin_bit_cast(unsigned int, x);
    u += 0x7FFFu + ((u >> 16) & 1u);
    return (unsigned short)(u >> 16);
}
__device__ __forceinline__ float bf16_to_f32(unsigned short h) {
    unsigned int u = ((unsigned int)h) << 16;
    return __builtin_bit_cast(float, u);
}

// 16x32 bf16 A-fragment (CDNA5 layout): lane (m = lane&15, ksel = lane>>4)
// holds K = [ksel*8 .. +7] and [ksel*8+16 .. +23]
__device__ __forceinline__ v16bf load_frag_a(const unsigned short* p) {
    union { v16bf v; u32x4 q[2]; } u;
    u.q[0] = *(const u32x4*)(p);
    u.q[1] = *(const u32x4*)(p + 16);
    return u.v;
}
// 32x16 bf16 B-fragment: lane (n = lane&15, ksel = lane>>4) holds
// K = [ksel*8 .. +7] and [ksel*8+16 .. +23] of column n, B transposed in LDS.
__device__ __forceinline__ v16bf load_frag_b(const unsigned short* p) {
    union { v16bf v; u32x4 q[2]; } u;
    u.q[0] = *(const u32x4*)(p);
    u.q[1] = *(const u32x4*)(p + 16);
    return u.v;
}

// ---------------------------------------------------------------------------
// Router: logits -> top-2 -> renormalized combine weights [T, 8]
// One wave32 per token.
// ---------------------------------------------------------------------------
__global__ __launch_bounds__(256)
void hy_router_kernel(const float* __restrict__ x,
                      const float* __restrict__ wg,
                      float* __restrict__ combine) {
    const int wave = threadIdx.x >> 5;
    const int lane = threadIdx.x & 31;
    const int t = blockIdx.x * 8 + wave;

    float acc[NEXP];
#pragma unroll
    for (int e = 0; e < NEXP; ++e) acc[e] = 0.0f;

    const float* xr = x + (size_t)t * HDIM;
    for (int h = lane; h < HDIM; h += 32) {
        const float xv = xr[h];
        const float* wr = wg + (size_t)h * NEXP;
#pragma unroll
        for (int e = 0; e < NEXP; ++e) acc[e] += xv * wr[e];
    }
#pragma unroll
    for (int e = 0; e < NEXP; ++e) {
#pragma unroll
        for (int off = 16; off > 0; off >>= 1)
            acc[e] += __shfl_xor(acc[e], off, 32);
    }
    if (lane == 0) {
        int i1 = 0;
#pragma unroll
        for (int e = 1; e < NEXP; ++e) if (acc[e] > acc[i1]) i1 = e;
        int i2 = (i1 == 0) ? 1 : 0;
#pragma unroll
        for (int e = 0; e < NEXP; ++e)
            if (e != i1 && acc[e] > acc[i2]) i2 = e;
        // softmax over top-2 logits == renormalized top-2 of full softmax
        const float p2 = __expf(acc[i2] - acc[i1]);
        const float inv = 1.0f / (1.0f + p2);
        float* cr = combine + (size_t)t * NEXP;
#pragma unroll
        for (int e = 0; e < NEXP; ++e)
            cr[e] = (e == i1) ? inv : ((e == i2) ? p2 * inv : 0.0f);
    }
}

// ---------------------------------------------------------------------------
// Tiled GEMM, fp32 in/out, bf16x3 split (hi*hi + hi*lo + lo*hi) via
// v_wmma_f32_16x16x32_bf16. Block tile 128x128, 8 waves, each wave 32x64.
// Register-prefetched K pipeline: next chunk's global loads are issued while
// the current chunk's WMMAs run out of LDS.
// Optional per-row A scale (strided) and optional C accumulate.
// M = gridDim.y*128, N = gridDim.x*128; K multiple of 32.
// ---------------------------------------------------------------------------
#define LDS_STRIDE 40  // ushort elems/row; 80B keeps 16B alignment, spreads banks

__global__ __launch_bounds__(256)
void hy_gemm_bf16x3_kernel(const float* __restrict__ A, int lda,
                           const float* __restrict__ B, int ldb,
                           float* __restrict__ C, int ldc,
                           int K,
                           const float* __restrict__ rowscale, int sstride,
                           int accumulate) {
    __shared__ unsigned short lsA[2][128 * LDS_STRIDE]; // hi/lo planes, [m][k]
    __shared__ unsigned short lsB[2][128 * LDS_STRIDE]; // hi/lo planes, [n][k]

    const int tid  = threadIdx.x;
    const int lane = tid & 31;
    const int wave = tid >> 5;
    const int wm = wave & 3;   // m offset 32*wm
    const int wn = wave >> 2;  // n offset 64*wn
    const int lh = lane & 15;
    const int ls = lane >> 4;

    const int bm = blockIdx.y * 128;
    const int bn = blockIdx.x * 128;

    // A loader: 8 threads x float4 cover one 32-col row; 32 rows/pass, 4 passes
    const int arow = tid >> 3;
    const int acol = (tid & 7) << 2;
    // B loader: 32 threads x float4 cover one 128-col row; 8 k-rows/pass, 4 passes
    const int bcol = (tid & 31) << 2;
    const int bk   = tid >> 5;

    // Per-thread row scales for the 4 A rows this thread stages (K-invariant).
    float ascale[4];
#pragma unroll
    for (int p = 0; p < 4; ++p) {
        const size_t gr = (size_t)(bm + arow + 32 * p);
        ascale[p] = rowscale ? rowscale[gr * (size_t)sstride] : 1.0f;
    }

    v8f c[2][4];
#pragma unroll
    for (int mt = 0; mt < 2; ++mt)
#pragma unroll
        for (int nt = 0; nt < 4; ++nt) {
            v8f z = {0.f, 0.f, 0.f, 0.f, 0.f, 0.f, 0.f, 0.f};
            c[mt][nt] = z;
        }

    // ---- prologue: load K-chunk 0 into registers ----
    f32x4 aR[4], bR[4];
#pragma unroll
    for (int p = 0; p < 4; ++p) {
        aR[p] = *(const f32x4*)(A + (size_t)(bm + arow + 32 * p) * lda + acol) * ascale[p];
        bR[p] = *(const f32x4*)(B + (size_t)(bk + 8 * p) * ldb + bn + bcol);
    }

    for (int kb = 0; kb < K; kb += 32) {
        // ---- stage current registers into LDS as bf16 hi/lo planes ----
#pragma unroll
        for (int p = 0; p < 4; ++p) {
            const int r = arow + 32 * p;
#pragma unroll
            for (int j = 0; j < 4; ++j) {
                const unsigned short h = f32_to_bf16(aR[p][j]);
                lsA[0][r * LDS_STRIDE + acol + j] = h;
                lsA[1][r * LDS_STRIDE + acol + j] =
                    f32_to_bf16(aR[p][j] - bf16_to_f32(h));
            }
            const int k = bk + 8 * p;
#pragma unroll
            for (int j = 0; j < 4; ++j) {
                const unsigned short h = f32_to_bf16(bR[p][j]);
                lsB[0][(bcol + j) * LDS_STRIDE + k] = h;
                lsB[1][(bcol + j) * LDS_STRIDE + k] =
                    f32_to_bf16(bR[p][j] - bf16_to_f32(h));
            }
        }
        __syncthreads();

        // ---- prefetch next K-chunk into registers (overlaps WMMA below) ----
        const int kn = kb + 32;
        if (kn < K) {
#pragma unroll
            for (int p = 0; p < 4; ++p) {
                aR[p] = *(const f32x4*)(A + (size_t)(bm + arow + 32 * p) * lda + kn + acol) *
                        ascale[p];
                bR[p] = *(const f32x4*)(B + (size_t)(kn + bk + 8 * p) * ldb + bn + bcol);
            }
            // deep prefetch (L2) two chunks ahead
            if (kn + 32 < K)
                __builtin_prefetch(A + (size_t)(bm + arow) * lda + kn + 32 + acol, 0, 1);
        }

        // ---- fragments out of LDS ----
        v16bf a_hi[2], a_lo[2], b_hi[4], b_lo[4];
#pragma unroll
        for (int mt = 0; mt < 2; ++mt) {
            const int row  = wm * 32 + mt * 16 + lh;
            const int base = row * LDS_STRIDE + ls * 8;
            a_hi[mt] = load_frag_a(&lsA[0][base]);
            a_lo[mt] = load_frag_a(&lsA[1][base]);
        }
#pragma unroll
        for (int nt = 0; nt < 4; ++nt) {
            const int n    = wn * 64 + nt * 16 + lh;
            const int base = n * LDS_STRIDE + ls * 8;
            b_hi[nt] = load_frag_b(&lsB[0][base]);
            b_lo[nt] = load_frag_b(&lsB[1][base]);
        }

        // ---- 2x4 output tiles x 3 split terms = 24 WMMA per K-chunk ----
#pragma unroll
        for (int mt = 0; mt < 2; ++mt) {
#pragma unroll
            for (int nt = 0; nt < 4; ++nt) {
                c[mt][nt] = __builtin_amdgcn_wmma_f32_16x16x32_bf16(
                    false, a_hi[mt], false, b_hi[nt], (short)0, c[mt][nt], false, false);
                c[mt][nt] = __builtin_amdgcn_wmma_f32_16x16x32_bf16(
                    false, a_hi[mt], false, b_lo[nt], (short)0, c[mt][nt], false, false);
                c[mt][nt] = __builtin_amdgcn_wmma_f32_16x16x32_bf16(
                    false, a_lo[mt], false, b_hi[nt], (short)0, c[mt][nt], false, false);
            }
        }
        __syncthreads();
    }

    // ---- writeback: C/D layout -> lane(n = lh), VGPR r -> m = r + ls*8 ----
#pragma unroll
    for (int mt = 0; mt < 2; ++mt) {
#pragma unroll
        for (int nt = 0; nt < 4; ++nt) {
            const int n = bn + wn * 64 + nt * 16 + lh;
#pragma unroll
            for (int r = 0; r < 8; ++r) {
                const int m = bm + wm * 32 + mt * 16 + ls * 8 + r;
                const size_t idx = (size_t)m * ldc + n;
                const float v = c[mt][nt][r];
                C[idx] = accumulate ? (C[idx] + v) : v;
            }
        }
    }
}

// ---------------------------------------------------------------------------
// act = silu(g) * u, from gu = [rows, 2*inter] (g | u), vectorized x4
// ---------------------------------------------------------------------------
__global__ __launch_bounds__(256)
void hy_silu_mul_kernel(const float* __restrict__ gu, float* __restrict__ act,
                        int rows, int inter) {
    const size_t n4 = (size_t)rows * inter / 4;
    const int i4 = inter / 4;
    for (size_t i = (size_t)blockIdx.x * 256 + threadIdx.x; i < n4;
         i += (size_t)gridDim.x * 256) {
        const size_t r = i / i4;
        const size_t c4 = i - r * i4;
        const f32x4 g = *(const f32x4*)(gu + r * 2 * inter + c4 * 4);
        const f32x4 u = *(const f32x4*)(gu + r * 2 * inter + inter + c4 * 4);
        f32x4 o;
#pragma unroll
        for (int j = 0; j < 4; ++j)
            o[j] = (g[j] / (1.0f + __expf(-g[j]))) * u[j];
        *(f32x4*)(act + i * 4) = o;
    }
}

// ---------------------------------------------------------------------------
// Launch
// ---------------------------------------------------------------------------
extern "C" void kernel_launch(void* const* d_in, const int* in_sizes, int n_in,
                              void* d_out, int out_size, void* d_ws, size_t ws_size,
                              hipStream_t stream) {
    (void)in_sizes; (void)n_in; (void)out_size; (void)ws_size;

    const float* x         = (const float*)d_in[0]; // [2,1024,2048] -> [2048,2048]
    const float* w_gate    = (const float*)d_in[1]; // [2048, 8]
    const float* w_gu      = (const float*)d_in[2]; // [8, 2048, 2048]
    const float* w_down    = (const float*)d_in[3]; // [8, 1024, 2048]
    const float* w_sh_gu   = (const float*)d_in[4]; // [2048, 8192]
    const float* w_sh_down = (const float*)d_in[5]; // [4096, 2048]
    float* out = (float*)d_out;                     // [2048, 2048]

    // workspace layout
    float* combine = (float*)d_ws;                   // 2048*8
    float* gu      = combine + (size_t)T_TOK * NEXP; // up to 2048*8192
    float* act     = gu + (size_t)T_TOK * 8192;      // up to 2048*4096

    const dim3 blk(256);

    // 1) router -> combine weights
    hy_router_kernel<<<T_TOK / 8, blk, 0, stream>>>(x, w_gate, combine);

    // 2) shared gate_up: [2048,2048]@[2048,8192] -> gu
    hy_gemm_bf16x3_kernel<<<dim3(8192 / 128, T_TOK / 128), blk, 0, stream>>>(
        x, HDIM, w_sh_gu, 8192, gu, 8192, HDIM, nullptr, 0, 0);
    // 3) act_sh = silu(g)*u  [2048,4096]
    hy_silu_mul_kernel<<<dim3(4096), blk, 0, stream>>>(gu, act, T_TOK, 4096);
    // 4) shared down: [2048,4096]@[4096,2048] -> out (overwrite)
    hy_gemm_bf16x3_kernel<<<dim3(HDIM / 128, T_TOK / 128), blk, 0, stream>>>(
        act, 4096, w_sh_down, HDIM, out, HDIM, 4096, nullptr, 0, 0);

    // 5) routed experts, sequential accumulation into out (stream-ordered)
    for (int e = 0; e < NEXP; ++e) {
        const float* wgu_e = w_gu + (size_t)e * HDIM * 2048;   // [2048, 2048]
        const float* wd_e  = w_down + (size_t)e * 1024 * HDIM; // [1024, 2048]
        // gate_up: [2048,2048]@[2048,2048] -> gu
        hy_gemm_bf16x3_kernel<<<dim3(2048 / 128, T_TOK / 128), blk, 0, stream>>>(
            x, HDIM, wgu_e, 2048, gu, 2048, HDIM, nullptr, 0, 0);
        // act_e = silu(g)*u  [2048,1024]
        hy_silu_mul_kernel<<<dim3(1024), blk, 0, stream>>>(gu, act, T_TOK, 1024);
        // down with combine[t,e] folded into A rows, accumulate into out
        hy_gemm_bf16x3_kernel<<<dim3(HDIM / 128, T_TOK / 128), blk, 0, stream>>>(
            act, 1024, wd_e, HDIM, out, HDIM, 1024, combine + e, NEXP, 1);
    }
}
